// QMixer_49452253446773
// MI455X (gfx1250) — matrix operand order repaired and checked by reading, not compiled, not run on hardware
//
#include <hip/hip_runtime.h>

typedef __attribute__((ext_vector_type(16))) _Float16 v16h;
typedef __attribute__((ext_vector_type(8)))  _Float16 v8h;
typedef __attribute__((ext_vector_type(4)))  _Float16 v4h;
typedef __attribute__((ext_vector_type(8)))  float    v8f;
typedef __attribute__((ext_vector_type(4)))  float    v4f;

#define NAg   8
#define NEn   8
#define NENT  16
#define NF    32
#define DD    64
#define AA    64
#define EE    32
#define STATE_F 512           // NAg*NF + NEn*NF
#define WAVES 4
#define NTHREADS 128
#define B_PER_WG 8            // WAVES * 2

#define WMMA_F16(Aop, Bop, Cop) \
  __builtin_amdgcn_wmma_f32_16x16x32_f16(false, (Aop), false, (Bop), (short)0, (Cop), false, false)

__device__ __forceinline__ v16h make_v16(v8h lo, v8h hi) {
  return __builtin_shufflevector(lo, hi, 0, 1, 2, 3, 4, 5, 6, 7, 8, 9, 10, 11, 12, 13, 14, 15);
}

// ---- A operand (16x32 f16) from row-major [16][ld]: per lane two contiguous
//      16B runs: K = k0+kh .. +7  and  K = k0+16+kh .. +7 ----
__device__ __forceinline__ v16h wmma_load_a(const _Float16* base, int ld, int k0, int lane) {
  const int row = lane & 15;
  const int kh  = (lane >> 4) << 3;
  const _Float16* p = base + row * ld + k0 + kh;
  return make_v16(*(const v8h*)p, *(const v8h*)(p + 16));
}

// ---- A operand from staged fp32 states (rows 0-7 = b0 entities, 8-15 = b1) ----
__device__ __forceinline__ v16h wmma_load_a_states(const float* st2b, int featbase, int lane) {
  const int row = lane & 15;
  const int kh  = (lane >> 4) << 3;
  const float* src = st2b + (row >> 3) * STATE_F + featbase + (row & 7) * NF + kh;
  v4f f0 = *(const v4f*)(src);
  v4f f1 = *(const v4f*)(src + 4);
  v4f f2 = *(const v4f*)(src + 16);
  v4f f3 = *(const v4f*)(src + 20);
  v16h a;
#pragma unroll
  for (int i = 0; i < 4; ++i) {
    a[i]      = (_Float16)f0[i];
    a[4 + i]  = (_Float16)f1[i];
    a[8 + i]  = (_Float16)f2[i];
    a[12 + i] = (_Float16)f3[i];
  }
  return a;
}

// ---- A operand for emi = [attn_out(64) || ally_embed(64)], rows = b*8+agent ----
__device__ __forceinline__ v16h wmma_load_a_emi(const _Float16* aoh /*[2][64]*/,
                                                const _Float16* ent /*[32][64]*/,
                                                int k0, int lane) {
  const int row = lane & 15;
  const int kh  = (lane >> 4) << 3;
  const int b   = row >> 3;
  const int ag  = row & 7;
  const _Float16* base = (k0 < DD) ? (aoh + b * DD + k0)
                                   : (ent + (b * 16 + ag) * DD + (k0 - DD));
  const _Float16* p = base + kh;
  return make_v16(*(const v8h*)p, *(const v8h*)(p + 16));
}

// ---- A operand: rows 0-7 = ao[b0], rows 8-15 = ao[b1] (row-replicated) so
//      C row (mb*8) -> c[0] holds this half-wave's own b ----
__device__ __forceinline__ v16h wmma_load_a_aorep(const _Float16* aoh /*[2][64]*/,
                                                  int k0, int lane) {
  const int b  = (lane & 15) >> 3;
  const int kh = (lane >> 4) << 3;
  const _Float16* p = aoh + b * DD + k0 + kh;
  return make_v16(*(const v8h*)p, *(const v8h*)(p + 16));
}

// ---- B operand from operand-packed weights: [chunk][lane][16 halfs] ----
__device__ __forceinline__ v16h wmma_load_bp(const _Float16* pk, int chunk, int lane) {
  const _Float16* p = pk + (((chunk << 5) + lane) << 4);
  return make_v16(*(const v8h*)p, *(const v8h*)(p + 8));
}

// ---- B operand = X^T, X row-major [n][k]: per lane K = k0+kh..+7 and k0+16+kh..+7 ----
__device__ __forceinline__ v16h wmma_load_bt(const _Float16* x, int ld, int k0, int lane) {
  const int col = lane & 15;
  const int kh  = (lane >> 4) << 3;
  const _Float16* p = x + col * ld + k0 + kh;
  return make_v16(*(const v8h*)p, *(const v8h*)(p + 16));
}

// ---- pack row-major fp32 weight [KTOT][NTOT] into B-operand layout ----
// coalesced global reads; chunk = (K/32)*(NTOT/16) + n/16 ;
// lane = (K/8 % 2)*16 + n%16 ; element = K%8 + 8*(K/16 % 2) ; optional constant scale
template <int KTOT, int NTOT>
__device__ __forceinline__ void pack_weight_b(_Float16* dst, const float* __restrict__ W,
                                              int tid, float scale = 1.0f) {
  constexpr int nChunks = NTOT >> 4;
  constexpr int total = KTOT * NTOT;
#pragma unroll 1
  for (int i = tid; i < total; i += NTHREADS) {
    const int K = i / NTOT;
    const int n = i - K * NTOT;
    const int chunk = (K >> 5) * nChunks + (n >> 4);
    const int ln = (((K >> 3) & 1) << 4) | (n & 15);
    const int e = (K & 7) | (((K >> 4) & 1) << 3);
    dst[(((chunk << 5) + ln) << 4) + e] = (_Float16)(W[i] * scale);
  }
}

// ---- store C (16x16 f32) to f16 row-major with bias; rows 0-7 -> row_lo,
//      rows 8-15 -> row_hi ----
__device__ __forceinline__ void wmma_store_f16(_Float16* dst, int ld, int row_lo, int row_hi,
                                               int n0, v8f c, const float* bias, int lane) {
  const int n  = lane & 15;
  const int rb = (lane >> 4) ? row_hi : row_lo;
  const float bv = bias[n0 + n];
#pragma unroll
  for (int r = 0; r < 8; ++r)
    dst[(rb + r) * ld + n0 + n] = (_Float16)(c[r] + bv);
}

__global__ __launch_bounds__(NTHREADS)
void qmix_fused_kernel(const float* __restrict__ agent_qs, const float* __restrict__ states,
                       const float* __restrict__ w_ally, const float* __restrict__ b_ally,
                       const float* __restrict__ w_enemy, const float* __restrict__ b_enemy,
                       const float* __restrict__ w_q, const float* __restrict__ b_q,
                       const float* __restrict__ w_k, const float* __restrict__ b_k,
                       const float* __restrict__ w_h1, const float* __restrict__ b_h1,
                       const float* __restrict__ w_hf, const float* __restrict__ b_hf,
                       const float* __restrict__ w_b1, const float* __restrict__ b_b1,
                       const float* __restrict__ w_v1, const float* __restrict__ b_v1,
                       const float* __restrict__ w_v2, const float* __restrict__ b_v2,
                       float* __restrict__ out, int nB)
{
  // operand-packed f16 weights
  __shared__ __align__(16) _Float16 s_wa[NF * DD];
  __shared__ __align__(16) _Float16 s_we[NF * DD];
  __shared__ __align__(16) _Float16 s_wq[DD * AA];   // pre-scaled by 1/8
  __shared__ __align__(16) _Float16 s_wk[DD * AA];
  __shared__ __align__(16) _Float16 s_wh1[2 * DD * EE];
  __shared__ __align__(16) _Float16 s_wb1p[DD * EE];
  __shared__ __align__(16) _Float16 s_whfp[DD * EE];
  __shared__ __align__(16) _Float16 s_wv1p[DD * EE];
  // per-wave intermediates
  __shared__ __align__(16) _Float16 s_ent[WAVES][32 * DD];
  __shared__ __align__(16) _Float16 s_q[WAVES][32 * AA];
  __shared__ __align__(16) _Float16 s_kk[WAVES][32 * AA];
  __shared__ __align__(16) _Float16 s_aoh[WAVES][2 * DD];
  __shared__ __align__(16) float    s_ao[WAVES][2 * DD];
  __shared__ float    s_rs[WAVES][2 * NENT];
  // staged inputs + small biases
  __shared__ __align__(16) float s_states[B_PER_WG][STATE_F];
  __shared__ float s_aq[B_PER_WG][NAg];
  __shared__ float s_ba[DD], s_be[DD], s_bq[AA], s_bk[AA];
  __shared__ float s_bh1[EE], s_bb1[EE], s_bhf[EE], s_bv1[EE], s_wv2[EE];

  const int tid  = threadIdx.x;
  const int lane = tid & 31;
  const int wave = tid >> 5;
  const int mb   = lane >> 4;
  const int baseB = blockIdx.x * B_PER_WG;

  // ---- cooperative: pack weights to operand layout, stage biases + inputs ----
  pack_weight_b<NF, DD>(s_wa, w_ally, tid);
  pack_weight_b<NF, DD>(s_we, w_enemy, tid);
  pack_weight_b<DD, AA>(s_wq, w_q, tid, 0.125f);   // fold 1/sqrt(A) into Wq
  pack_weight_b<DD, AA>(s_wk, w_k, tid);
  pack_weight_b<2 * DD, EE>(s_wh1, w_h1, tid);
  pack_weight_b<DD, EE>(s_wb1p, w_b1, tid);
  pack_weight_b<DD, EE>(s_whfp, w_hf, tid);
  pack_weight_b<DD, EE>(s_wv1p, w_v1, tid);
  for (int i = tid; i < DD; i += NTHREADS) {
    s_ba[i] = b_ally[i]; s_be[i] = b_enemy[i];
    s_bq[i] = b_q[i] * 0.125f;                     // fold 1/sqrt(A) into bq
    s_bk[i] = b_k[i];
  }
  for (int i = tid; i < EE; i += NTHREADS) {
    s_bh1[i] = b_h1[i]; s_bb1[i] = b_b1[i]; s_bhf[i] = b_hf[i];
    s_bv1[i] = b_v1[i]; s_wv2[i] = w_v2[i];
  }
  {
    const float* src = states + (size_t)baseB * STATE_F;
    if (baseB + B_PER_WG <= nB) {
      const v4f* s4 = (const v4f*)src;
      v4f* d4 = (v4f*)&s_states[0][0];
#pragma unroll 1
      for (int i = tid; i < (B_PER_WG * STATE_F) / 4; i += NTHREADS) d4[i] = s4[i];
    } else {
      const int nvalid = (nB - baseB) * STATE_F;
#pragma unroll 1
      for (int i = tid; i < B_PER_WG * STATE_F; i += NTHREADS)
        s_states[0][i] = (i < nvalid) ? src[i] : 0.0f;
    }
    const float* qsrc = agent_qs + (size_t)baseB * NAg;
    const int qvalid = (nB - baseB < B_PER_WG ? nB - baseB : B_PER_WG) * NAg;
    for (int i = tid; i < B_PER_WG * NAg; i += NTHREADS)
      s_aq[0][i] = (i < qvalid) ? qsrc[i] : 0.0f;
  }
  __syncthreads();

  const float* st = &s_states[wave * 2][0];

  // ======== phase 1: entity embeddings (ally + enemy), 8 WMMA ========
  {
    v16h aAl = wmma_load_a_states(st, 0, lane);
    v16h aEn = wmma_load_a_states(st, NAg * NF, lane);
#pragma unroll
    for (int nc = 0; nc < 4; ++nc) {
      v8f c0 = {};
      c0 = WMMA_F16(aAl, wmma_load_bp(s_wa, nc, lane), c0);
      wmma_store_f16(&s_ent[wave][0], DD, 0, 16, nc * 16, c0, s_ba, lane);
      v8f c1 = {};
      c1 = WMMA_F16(aEn, wmma_load_bp(s_we, nc, lane), c1);
      wmma_store_f16(&s_ent[wave][0], DD, NAg, 16 + NAg, nc * 16, c1, s_be, lane);
    }
  }
  __syncthreads();

  // ======== phase 2: Q and K projections, 32 WMMA ========
  // nc outer / tile inner: each packed B chunk is loaded exactly once.
  {
    v16h ea[2][2];   // [tile][kstep]
#pragma unroll
    for (int tile = 0; tile < 2; ++tile) {
      const _Float16* eb = &s_ent[wave][tile * 16 * DD];
      ea[tile][0] = wmma_load_a(eb, DD, 0, lane);
      ea[tile][1] = wmma_load_a(eb, DD, 32, lane);
    }
#pragma unroll
    for (int nc = 0; nc < 4; ++nc) {
      const v16h bq0 = wmma_load_bp(s_wq, 0 * 4 + nc, lane);
      const v16h bq1 = wmma_load_bp(s_wq, 1 * 4 + nc, lane);
      const v16h bk0 = wmma_load_bp(s_wk, 0 * 4 + nc, lane);
      const v16h bk1 = wmma_load_bp(s_wk, 1 * 4 + nc, lane);
#pragma unroll
      for (int tile = 0; tile < 2; ++tile) {
        v8f cq = {};
        cq = WMMA_F16(ea[tile][0], bq0, cq);
        cq = WMMA_F16(ea[tile][1], bq1, cq);
        wmma_store_f16(&s_q[wave][0], AA, tile * 16, tile * 16 + 8, nc * 16, cq, s_bq, lane);
        v8f ck = {};
        ck = WMMA_F16(ea[tile][0], bk0, ck);
        ck = WMMA_F16(ea[tile][1], bk1, ck);
        wmma_store_f16(&s_kk[wave][0], AA, tile * 16, tile * 16 + 8, nc * 16, ck, s_bk, lane);
      }
    }
  }
  __syncthreads();

  // ======== phase 3: energy = Q K^T, softmax over query axis, row-sums ========
#pragma unroll
  for (int b = 0; b < 2; ++b) {
    const _Float16* qb = &s_q[wave][b * 16 * AA];
    const _Float16* kb = &s_kk[wave][b * 16 * AA];
    v8f ce = {};
    ce = WMMA_F16(wmma_load_a(qb, AA, 0, lane),  wmma_load_bt(kb, AA, 0, lane),  ce);
    ce = WMMA_F16(wmma_load_a(qb, AA, 32, lane), wmma_load_bt(kb, AA, 32, lane), ce);
    // column (M-axis) softmax: my 8 rows + partner half via xor-16
    float mx = ce[0];
#pragma unroll
    for (int r = 1; r < 8; ++r) mx = fmaxf(mx, ce[r]);
    mx = fmaxf(mx, __shfl_xor(mx, 16, 32));
    float ev[8]; float ps = 0.0f;
#pragma unroll
    for (int r = 0; r < 8; ++r) { ev[r] = __expf(ce[r] - mx); ps += ev[r]; }
    const float tot = ps + __shfl_xor(ps, 16, 32);
    const float inv = 1.0f / tot;
#pragma unroll
    for (int r = 0; r < 8; ++r) {
      float v = ev[r] * inv;
      v += __shfl_xor(v, 1, 32); v += __shfl_xor(v, 2, 32);
      v += __shfl_xor(v, 4, 32); v += __shfl_xor(v, 8, 32);
      if ((lane & 15) == 0) s_rs[wave][b * NENT + mb * 8 + r] = v;
    }
  }
  __syncthreads();

  // ======== phase 4: attn_out[d] = (1/16) sum_n rowsum[n] * entity[n][d] ========
  {
    const int b  = lane >> 4;          // b handled by this half-wave
    const int d0 = (lane & 15) * 4;    // 4 consecutive d per lane
    float acc0 = 0.f, acc1 = 0.f, acc2 = 0.f, acc3 = 0.f;
#pragma unroll
    for (int n = 0; n < NENT; ++n) {
      const float w = s_rs[wave][b * NENT + n];
      v4h ev = *(const v4h*)&s_ent[wave][(b * 16 + n) * DD + d0];
      acc0 += w * (float)ev[0];
      acc1 += w * (float)ev[1];
      acc2 += w * (float)ev[2];
      acc3 += w * (float)ev[3];
    }
    acc0 *= (1.0f / 16.0f); acc1 *= (1.0f / 16.0f);
    acc2 *= (1.0f / 16.0f); acc3 *= (1.0f / 16.0f);
    v4f af = {acc0, acc1, acc2, acc3};
    *(v4f*)&s_ao[wave][b * DD + d0] = af;
    v4h ah = {(_Float16)acc0, (_Float16)acc1, (_Float16)acc2, (_Float16)acc3};
    *(v4h*)&s_aoh[wave][b * DD + d0] = ah;
  }
  __syncthreads();

  // ======== phase 5: W1 = |emi @ w_h1 + b_h1|, 8 WMMA ========
  v16h aemi[4];
#pragma unroll
  for (int ks = 0; ks < 4; ++ks)
    aemi[ks] = wmma_load_a_emi(&s_aoh[wave][0], &s_ent[wave][0], ks * 32, lane);

  float w1v[2][8];
#pragma unroll
  for (int t = 0; t < 2; ++t) {
    v8f c = {};
#pragma unroll
    for (int ks = 0; ks < 4; ++ks)
      c = WMMA_F16(aemi[ks], wmma_load_bp(s_wh1, ks * 2 + t, lane), c);
    const float bh = s_bh1[t * 16 + (lane & 15)];
#pragma unroll
    for (int r = 0; r < 8; ++r) w1v[t][r] = fabsf(c[r] + bh);
  }

  // ======== phase 6: hypernet matvecs as WMMA (b1, w_final, v1), 12 WMMA ========
  // A rows replicate: rows 0-7 = ao[b0], rows 8-15 = ao[b1]  =>  c[0] on each
  // lane is the result for that half-wave's own b. No shuffles needed.
  const v16h aoA0 = wmma_load_a_aorep(&s_aoh[wave][0], 0, lane);
  const v16h aoA1 = wmma_load_a_aorep(&s_aoh[wave][0], 32, lane);

  float aqv[8];
#pragma unroll
  for (int r = 0; r < 8; ++r) aqv[r] = s_aq[wave * 2 + mb][r];

  float ypart = 0.0f;
#pragma unroll
  for (int t = 0; t < 2; ++t) {
    const int e = t * 16 + (lane & 15);
    v8f cb = {};
    cb = WMMA_F16(aoA0, wmma_load_bp(s_wb1p, 0 + t, lane), cb);
    cb = WMMA_F16(aoA1, wmma_load_bp(s_wb1p, 2 + t, lane), cb);
    v8f cf = {};
    cf = WMMA_F16(aoA0, wmma_load_bp(s_whfp, 0 + t, lane), cf);
    cf = WMMA_F16(aoA1, wmma_load_bp(s_whfp, 2 + t, lane), cf);
    v8f cv = {};
    cv = WMMA_F16(aoA0, wmma_load_bp(s_wv1p, 0 + t, lane), cv);
    cv = WMMA_F16(aoA1, wmma_load_bp(s_wv1p, 2 + t, lane), cv);

    const float b1 = cb[0] + s_bb1[e];
    const float wf = fabsf(cf[0] + s_bhf[e]);
    const float v1 = cv[0] + s_bv1[e];

    float contrib = 0.0f;
#pragma unroll
    for (int r = 0; r < 8; ++r) contrib += aqv[r] * w1v[t][r];
    const float x = contrib + b1;
    const float hid = (x > 0.0f) ? x : (__expf(x) - 1.0f);   // elu
    ypart += hid * wf + fmaxf(v1, 0.0f) * s_wv2[e];
  }

  ypart += __shfl_xor(ypart, 1, 32); ypart += __shfl_xor(ypart, 2, 32);
  ypart += __shfl_xor(ypart, 4, 32); ypart += __shfl_xor(ypart, 8, 32);

  const int bg = baseB + wave * 2 + mb;
  if ((lane & 15) == 0 && bg < nB) out[bg] = ypart + b_v2[0];
}

extern "C" void kernel_launch(void* const* d_in, const int* in_sizes, int n_in,
                              void* d_out, int out_size, void* d_ws, size_t ws_size,
                              hipStream_t stream) {
  const float* agent_qs = (const float*)d_in[0];
  const float* states   = (const float*)d_in[1];
  const float* w_ally   = (const float*)d_in[2];
  const float* b_ally   = (const float*)d_in[3];
  const float* w_enemy  = (const float*)d_in[4];
  const float* b_enemy  = (const float*)d_in[5];
  const float* w_q      = (const float*)d_in[6];
  const float* b_q      = (const float*)d_in[7];
  const float* w_k      = (const float*)d_in[8];
  const float* b_k      = (const float*)d_in[9];
  const float* w_h1     = (const float*)d_in[10];
  const float* b_h1     = (const float*)d_in[11];
  const float* w_hf     = (const float*)d_in[12];
  const float* b_hf     = (const float*)d_in[13];
  const float* w_b1     = (const float*)d_in[14];
  const float* b_b1     = (const float*)d_in[15];
  const float* w_v1     = (const float*)d_in[16];
  const float* b_v1     = (const float*)d_in[17];
  const float* w_v2     = (const float*)d_in[18];
  const float* b_v2     = (const float*)d_in[19];
  float* out = (float*)d_out;

  const int nB = in_sizes[0] / NAg;           // bs*sl = 65536
  const int grid = (nB + B_PER_WG - 1) / B_PER_WG;
  qmix_fused_kernel<<<grid, NTHREADS, 0, stream>>>(
      agent_qs, states, w_ally, b_ally, w_enemy, b_enemy, w_q, b_q, w_k, b_k,
      w_h1, b_h1, w_hf, b_hf, w_b1, b_b1, w_v1, b_v1, w_v2, b_v2, out, nB);
}
